// MultiQueryAttention_3779571220778
// MI455X (gfx1250) — matrix operand order repaired and checked by reading, not compiled, not run on hardware
//
#include <hip/hip_runtime.h>
#include <hip/hip_bf16.h>

// ---------------------------------------------------------------------------
// MQA block for MI455X (gfx1250, wave32, WMMA 16x16x32 f16 -> f32)
//   1) cvt X -> f16 ; transpose weights -> f16 [N][K]
//   2) WMMA GEMM: Qh = (Xh@Wq + bq) * 1/8   (f16, [4096][1024])
//                 Kh = (Xh@Wk + bk)         (f16, [4096][64])
//                 Vt = (Xh@Wv + bv)^T       (f16, [64][4096])
//   3) flash attention, 64-key chunks; K/V chunks double-buffered in LDS via
//      async global->LDS (shared by the 4 waves of a block = 4 query tiles of
//      the same (b,h))            -> Oh f16 [4096][1024]
//   4) WMMA GEMM: Y = Oh@Wo + bo  (f32 = d_out)
// ---------------------------------------------------------------------------

typedef _Float16 h8   __attribute__((ext_vector_type(8)));
typedef _Float16 v16h __attribute__((ext_vector_type(16)));
typedef float    v8f  __attribute__((ext_vector_type(8)));

#define BATCH   2
#define SEQ     2048
#define HID     1024
#define NHEAD   16
#define HDIM    64
#define ROWS    (BATCH * SEQ)   // 4096
#define CH      64              // keys per attention chunk
#define LDP     72              // padded LDS row stride (f16 elems)

#if defined(__gfx1250__) &&                                        \
    __has_builtin(__builtin_amdgcn_global_load_async_to_lds_b128) && \
    __has_builtin(__builtin_amdgcn_s_wait_asynccnt)
#define USE_ASYNC 1
#define AS1 __attribute__((address_space(1)))
#define AS3 __attribute__((address_space(3)))
typedef int v4i_g __attribute__((vector_size(16)));  // matches builtin param
#endif
#ifndef USE_ASYNC
#define USE_ASYNC 0
#endif

__device__ __forceinline__ v8f wmma_f16f32(v16h a, v16h b, v8f c) {
  return __builtin_amdgcn_wmma_f32_16x16x32_f16(
      /*neg_a=*/false, a, /*neg_b=*/false, b,
      /*c_mod=*/(short)0, c, /*reuse_a=*/false, /*reuse_b=*/false);
}

// A-fragment (16x32 f16): lane row m = lane%16; elems 0..15 -> K=half*16+i
// (half = lane>>4). Two 16B loads.
__device__ __forceinline__ v16h load_afrag(const _Float16* rowbase, int half) {
  h8 lo = *(const h8*)(rowbase + half * 16);
  h8 hi = *(const h8*)(rowbase + half * 16 + 8);
  v16h a;
#pragma unroll
  for (int i = 0; i < 8; ++i) { a[i] = lo[i]; a[i + 8] = hi[i]; }
  return a;
}

// B-fragment: lane holds column n = lane%16, K = half*16 + 0..15 contiguous.
// Two 16B loads so only 16B alignment is required (LDS rows stride 144B).
__device__ __forceinline__ v16h load_bfrag(const _Float16* p) {
  h8 lo = *(const h8*)p;
  h8 hi = *(const h8*)(p + 8);
  v16h v;
#pragma unroll
  for (int i = 0; i < 8; ++i) { v[i] = lo[i]; v[i + 8] = hi[i]; }
  return v;
}

// ---------------------------------------------------------------------------
// conversion / transpose helpers (run once per call, tiny cost)
// ---------------------------------------------------------------------------
__global__ void cvt_f32_f16(const float* __restrict__ src,
                            _Float16* __restrict__ dst, size_t n) {
  size_t i = (size_t)blockIdx.x * 256 + threadIdx.x;
  if (i < n) dst[i] = (_Float16)src[i];
}

__global__ void transpose_f32_f16(const float* __restrict__ src,
                                  _Float16* __restrict__ dst, int K, int N) {
  size_t idx = (size_t)blockIdx.x * 256 + threadIdx.x;
  if (idx >= (size_t)N * K) return;
  int k = (int)(idx % K);
  int n = (int)(idx / K);
  dst[idx] = (_Float16)src[(size_t)k * N + n];
}

// ---------------------------------------------------------------------------
// WMMA GEMM: C = (A[M][K] @ Bt[N][K]^T + bias) * scale ; wave tile 16x64
// OUT_MODE: 0 = f32 [M][N], 1 = f16 [M][N], 2 = f16 transposed [N][M]
// ---------------------------------------------------------------------------
template <int OUT_MODE>
__global__ __launch_bounds__(128) void gemm16x64(
    const _Float16* __restrict__ A, const _Float16* __restrict__ Bt,
    const float* __restrict__ bias, void* __restrict__ Cout,
    int M, int N, int K, float scale, int tiles_n) {
  const int wave = blockIdx.x * 4 + (threadIdx.x >> 5);
  const int lane = threadIdx.x & 31;
  const int n16  = lane & 15;
  const int half = lane >> 4;
  const int tm = wave / tiles_n;
  const int tn = wave % tiles_n;

  const _Float16* arow = A + (size_t)(tm * 16 + n16) * K;
  const _Float16* brow = Bt + (size_t)(tn * 64 + n16) * K + half * 16;

  v8f acc[4];
#pragma unroll
  for (int i = 0; i < 4; ++i) acc[i] = (v8f)0.0f;

  for (int kc = 0; kc < K; kc += 32) {
    v16h af = load_afrag(arow + kc, half);
    v16h b0 = *(const v16h*)(brow + kc);
    v16h b1 = *(const v16h*)(brow + kc + (size_t)16 * K);
    v16h b2 = *(const v16h*)(brow + kc + (size_t)32 * K);
    v16h b3 = *(const v16h*)(brow + kc + (size_t)48 * K);
    acc[0] = wmma_f16f32(af, b0, acc[0]);
    acc[1] = wmma_f16f32(af, b1, acc[1]);
    acc[2] = wmma_f16f32(af, b2, acc[2]);
    acc[3] = wmma_f16f32(af, b3, acc[3]);
  }

  // C layout: VGPR j -> row m = tm*16 + 8*half + j, col n = tn*64 + nt*16 + n16
#pragma unroll
  for (int nt = 0; nt < 4; ++nt) {
    const int n = tn * 64 + nt * 16 + n16;
    const float bs = bias[n];
    if (OUT_MODE == 2) {
      h8 o;
#pragma unroll
      for (int j = 0; j < 8; ++j) o[j] = (_Float16)((acc[nt][j] + bs) * scale);
      *(h8*)((_Float16*)Cout + (size_t)n * M + tm * 16 + 8 * half) = o;
    } else if (OUT_MODE == 1) {
      _Float16* c = (_Float16*)Cout;
#pragma unroll
      for (int j = 0; j < 8; ++j)
        c[(size_t)(tm * 16 + 8 * half + j) * N + n] =
            (_Float16)((acc[nt][j] + bs) * scale);
    } else {
      float* c = (float*)Cout;
#pragma unroll
      for (int j = 0; j < 8; ++j)
        c[(size_t)(tm * 16 + 8 * half + j) * N + n] = (acc[nt][j] + bs) * scale;
    }
  }
}

// ---------------------------------------------------------------------------
// Flash attention. Block = 4 waves = 4 query tiles of the same (b, head);
// they share each 64-key K/V chunk, staged in LDS (async, double-buffered).
// Qh f16 [4096][1024] (pre-scaled by 1/8); Kh f16 [4096][64];
// Vt f16 [64][4096]; amask int [B][S]; Oh f16 [4096][1024]
// ---------------------------------------------------------------------------
__global__ __launch_bounds__(128) void mqa_flash_attn(
    const _Float16* __restrict__ Qh, const _Float16* __restrict__ Kh,
    const _Float16* __restrict__ Vt, const int* __restrict__ amask,
    _Float16* __restrict__ Oh) {
  __shared__ _Float16 Kl[2][CH][LDP];   // [buf][key][d]   18 KB
  __shared__ _Float16 Vl[2][HDIM][LDP]; // [buf][d][key]   18 KB
  __shared__ _Float16 pP[4][16][LDP];   // per-wave P tile  9 KB

  const int tid  = threadIdx.x;
  const int w    = tid >> 5;
  const int lane = tid & 31;
  const int n16  = lane & 15;
  const int half = lane >> 4;
  const int wave = blockIdx.x * 4 + w;

  const int qt = wave & 127;          // query tile (fastest) -> block shares b,h
  const int hh = (wave >> 7) & 15;    // head
  const int b  = wave >> 11;          // batch
  const int brow = b * SEQ;

  // stage one 64-key chunk of K and Vt into LDS buffer `buf`
  auto issue_chunk = [&](int buf, int t0) {
#if USE_ASYNC
#pragma unroll
    for (int q = 0; q < 4; ++q) {      // 512 x 16B K segs over 128 threads
      int s = tid + q * 128, r = s >> 3, c = (s & 7) * 8;
      __builtin_amdgcn_global_load_async_to_lds_b128(
          (AS1 v4i_g*)(Kh + (size_t)(brow + t0 + r) * HDIM + c),
          (AS3 v4i_g*)&Kl[buf][r][c], 0, 0);
    }
#pragma unroll
    for (int q = 0; q < 4; ++q) {
      int s = tid + q * 128, r = s >> 3, c = (s & 7) * 8;
      __builtin_amdgcn_global_load_async_to_lds_b128(
          (AS1 v4i_g*)(Vt + (size_t)r * ROWS + brow + t0 + c),
          (AS3 v4i_g*)&Vl[buf][r][c], 0, 0);
    }
#else
#pragma unroll
    for (int q = 0; q < 4; ++q) {
      int s = tid + q * 128, r = s >> 3, c = (s & 7) * 8;
      *(h8*)&Kl[buf][r][c] = *(const h8*)(Kh + (size_t)(brow + t0 + r) * HDIM + c);
    }
#pragma unroll
    for (int q = 0; q < 4; ++q) {
      int s = tid + q * 128, r = s >> 3, c = (s & 7) * 8;
      *(h8*)&Vl[buf][r][c] = *(const h8*)(Vt + (size_t)r * ROWS + brow + t0 + c);
    }
#endif
  };

  // Q fragments (d = 0..31, 32..63), kept in registers throughout
  const _Float16* qbase = Qh + (size_t)(brow + qt * 16 + n16) * HID + hh * HDIM;
  v16h qa0 = load_afrag(qbase, half);
  v16h qa1 = load_afrag(qbase + 32, half);

  v8f acc[4];
#pragma unroll
  for (int i = 0; i < 4; ++i) acc[i] = (v8f)0.0f;
  float mrun[8], lrun[8];
#pragma unroll
  for (int j = 0; j < 8; ++j) { mrun[j] = -3.0e38f; lrun[j] = 0.0f; }

  constexpr int NCH = SEQ / CH;  // 32 chunks
#if USE_ASYNC
  issue_chunk(0, 0);
#endif
  for (int it = 0; it < NCH; ++it) {
    const int t0 = it * CH;
    const int buf = it & 1;
#if USE_ASYNC
    if (it + 1 < NCH) {
      issue_chunk(buf ^ 1, t0 + CH);            // prefetch next (8 asyncs/wave)
      __builtin_amdgcn_s_wait_asynccnt(8);      // current chunk complete
    } else {
      __builtin_amdgcn_s_wait_asynccnt(0);
    }
#else
    issue_chunk(buf, t0);
#endif
    __syncthreads();  // K/V chunk visible to all 4 waves

    // ---- scores: four 16x16 tiles over the 64 keys -------------------------
    v8f s[4];
#pragma unroll
    for (int tt = 0; tt < 4; ++tt) {
      s[tt] = (v8f)0.0f;
      const _Float16* krow = &Kl[buf][tt * 16 + n16][0];
      s[tt] = wmma_f16f32(qa0, load_bfrag(krow + half * 16), s[tt]);
      s[tt] = wmma_f16f32(qa1, load_bfrag(krow + 32 + half * 16), s[tt]);
    }

    int am[4];
#pragma unroll
    for (int tt = 0; tt < 4; ++tt) am[tt] = amask[brow + t0 + tt * 16 + n16];

    // ---- online softmax (row = 16 lanes of one half; 4 cols per lane) ------
    float alpha[8];
#pragma unroll
    for (int j = 0; j < 8; ++j) {
      float a0 = am[0] ? (float)s[0][j] : -3.0e38f;
      float a1 = am[1] ? (float)s[1][j] : -3.0e38f;
      float a2 = am[2] ? (float)s[2][j] : -3.0e38f;
      float a3 = am[3] ? (float)s[3][j] : -3.0e38f;
      float mt = fmaxf(fmaxf(a0, a1), fmaxf(a2, a3));
      mt = fmaxf(mt, __shfl_xor(mt, 1));
      mt = fmaxf(mt, __shfl_xor(mt, 2));
      mt = fmaxf(mt, __shfl_xor(mt, 4));
      mt = fmaxf(mt, __shfl_xor(mt, 8));
      const float mnew = fmaxf(mrun[j], mt);
      const float al = __expf(mrun[j] - mnew);
      const float p0 = __expf(a0 - mnew);
      const float p1 = __expf(a1 - mnew);
      const float p2 = __expf(a2 - mnew);
      const float p3 = __expf(a3 - mnew);
      float rs = (p0 + p1) + (p2 + p3);
      rs += __shfl_xor(rs, 1);
      rs += __shfl_xor(rs, 2);
      rs += __shfl_xor(rs, 4);
      rs += __shfl_xor(rs, 8);
      lrun[j] = lrun[j] * al + rs;
      mrun[j] = mnew;
      alpha[j] = al;
      s[0][j] = p0; s[1][j] = p1; s[2][j] = p2; s[3][j] = p3;
    }
#pragma unroll
    for (int nt = 0; nt < 4; ++nt)
#pragma unroll
      for (int j = 0; j < 8; ++j) acc[nt][j] *= alpha[j];

    // ---- P: C-layout -> per-wave LDS tile -> A-layout ----------------------
#pragma unroll
    for (int j = 0; j < 8; ++j) {
      const int row = 8 * half + j;
      pP[w][row][n16]      = (_Float16)s[0][j];
      pP[w][row][16 + n16] = (_Float16)s[1][j];
      pP[w][row][32 + n16] = (_Float16)s[2][j];
      pP[w][row][48 + n16] = (_Float16)s[3][j];
    }
    // same-wave LDS ops complete in order; no block barrier needed here
    v16h pa0 = load_afrag(&pP[w][n16][0], half);
    v16h pa1 = load_afrag(&pP[w][n16][32], half);

    // ---- out += P @ V ------------------------------------------------------
#pragma unroll
    for (int nt = 0; nt < 4; ++nt) {
      const _Float16* vrow = &Vl[buf][nt * 16 + n16][0];
      acc[nt] = wmma_f16f32(pa0, load_bfrag(vrow + half * 16), acc[nt]);
      acc[nt] = wmma_f16f32(pa1, load_bfrag(vrow + 32 + half * 16), acc[nt]);
    }
    __syncthreads();  // all waves done with buf before it is refilled
  }

  // ---- normalize and store (heads concatenated: col = h*64 + d) ------------
  _Float16* obase =
      Oh + (size_t)(brow + qt * 16 + 8 * half) * HID + hh * HDIM + n16;
#pragma unroll
  for (int j = 0; j < 8; ++j) {
    const float r = 1.0f / lrun[j];
    obase[(size_t)j * HID + 0]  = (_Float16)(acc[0][j] * r);
    obase[(size_t)j * HID + 16] = (_Float16)(acc[1][j] * r);
    obase[(size_t)j * HID + 32] = (_Float16)(acc[2][j] * r);
    obase[(size_t)j * HID + 48] = (_Float16)(acc[3][j] * r);
  }
}

// ---------------------------------------------------------------------------
extern "C" void kernel_launch(void* const* d_in, const int* in_sizes, int n_in,
                              void* d_out, int out_size, void* d_ws,
                              size_t ws_size, hipStream_t stream) {
  (void)in_sizes; (void)n_in; (void)out_size; (void)ws_size;
  const float* X    = (const float*)d_in[0];
  const float* Wq   = (const float*)d_in[1];
  const float* bq   = (const float*)d_in[2];
  const float* Wk   = (const float*)d_in[3];
  const float* bk   = (const float*)d_in[4];
  const float* Wv   = (const float*)d_in[5];
  const float* bv   = (const float*)d_in[6];
  const float* Wo   = (const float*)d_in[7];
  const float* bo   = (const float*)d_in[8];
  const int*   mask = (const int*)d_in[9];
  float* Y = (float*)d_out;

  char* ws = (char*)d_ws;
  size_t off = 0;
  auto alloc = [&](size_t bytes) -> void* {
    void* p = ws + off;
    off += (bytes + 255) & ~(size_t)255;
    return p;
  };
  _Float16* Xh  = (_Float16*)alloc((size_t)ROWS * HID * 2);
  _Float16* Wqt = (_Float16*)alloc((size_t)HID * HID * 2);
  _Float16* Wkt = (_Float16*)alloc((size_t)HDIM * HID * 2);
  _Float16* Wvt = (_Float16*)alloc((size_t)HDIM * HID * 2);
  _Float16* Wot = (_Float16*)alloc((size_t)HID * HID * 2);
  _Float16* Qh  = (_Float16*)alloc((size_t)ROWS * HID * 2);
  _Float16* Khb = (_Float16*)alloc((size_t)ROWS * HDIM * 2);
  _Float16* Vt  = (_Float16*)alloc((size_t)HDIM * ROWS * 2);
  _Float16* Oh  = (_Float16*)alloc((size_t)ROWS * HID * 2);

  // 1) precision conversion + weight transposes
  cvt_f32_f16<<<(ROWS * HID + 255) / 256, 256, 0, stream>>>(
      X, Xh, (size_t)ROWS * HID);
  transpose_f32_f16<<<(HID * HID + 255) / 256, 256, 0, stream>>>(Wq, Wqt, HID, HID);
  transpose_f32_f16<<<(HID * HDIM + 255) / 256, 256, 0, stream>>>(Wk, Wkt, HID, HDIM);
  transpose_f32_f16<<<(HID * HDIM + 255) / 256, 256, 0, stream>>>(Wv, Wvt, HID, HDIM);
  transpose_f32_f16<<<(HID * HID + 255) / 256, 256, 0, stream>>>(Wo, Wot, HID, HID);

  // 2) projections (Q carries the 1/sqrt(64) softmax scale)
  gemm16x64<1><<<(ROWS / 16) * (HID / 64) / 4, 128, 0, stream>>>(
      Xh, Wqt, bq, Qh, ROWS, HID, HID, 0.125f, HID / 64);
  gemm16x64<1><<<(ROWS / 16) * (HDIM / 64) / 4, 128, 0, stream>>>(
      Xh, Wkt, bk, Khb, ROWS, HDIM, HID, 1.0f, HDIM / 64);
  gemm16x64<2><<<(ROWS / 16) * (HDIM / 64) / 4, 128, 0, stream>>>(
      Xh, Wvt, bv, Vt, ROWS, HDIM, HID, 1.0f, HDIM / 64);

  // 3) flash attention: B*H*(S/16) = 4096 waves, 4 per block (same b,h)
  mqa_flash_attn<<<BATCH * NHEAD * (SEQ / 16) / 4, 128, 0, stream>>>(
      Qh, Khb, Vt, mask, Oh);

  // 4) output projection -> fp32 result
  gemm16x64<0><<<(ROWS / 16) * (HID / 64) / 4, 128, 0, stream>>>(
      Oh, Wot, bo, Y, ROWS, HID, HID, 1.0f, HID / 64);
}
